// MultiHeadedAttention_39986145526470
// MI455X (gfx1250) — matrix operand order repaired and checked by reading, not compiled, not run on hardware
//
#include <hip/hip_runtime.h>

typedef __attribute__((ext_vector_type(16))) __bf16 v16bf;
typedef __attribute__((ext_vector_type(8)))  __bf16 v8bf;
typedef __attribute__((ext_vector_type(8)))  float  v8f;

namespace {

constexpr int Bn  = 2;
constexpr int Sn  = 2048;
constexpr int En  = 1024;
constexpr int Hn  = 16;
constexpr int DKn = 64;
constexpr int Mn  = Bn * Sn;   // 4096 token rows

enum OutMode { MODE_HEAD = 0, MODE_VT = 1, MODE_F32 = 2 };

// ---- one-shot f32 -> bf16 conversion (8 elems/thread, vector ld/st) --------
__global__ void cvt_bf16_kernel(const float* __restrict__ in,
                                __bf16* __restrict__ out, int n8) {
  const int t = blockIdx.x * blockDim.x + threadIdx.x;
  if (t >= n8) return;
  const size_t base = (size_t)t * 8;
  float4 a = *(const float4*)(in + base);
  float4 b = *(const float4*)(in + base + 4);
  v8bf r;
  r[0] = (__bf16)a.x; r[1] = (__bf16)a.y; r[2] = (__bf16)a.z; r[3] = (__bf16)a.w;
  r[4] = (__bf16)b.x; r[5] = (__bf16)b.y; r[6] = (__bf16)b.z; r[7] = (__bf16)b.w;
  *(v8bf*)(out + base) = r;
}

// ---- fragment load helpers -------------------------------------------------
// A fragment (16x32 bf16): per lane one contiguous 16-elem K run: p[0..15]
__device__ inline v16bf load_afrag(const __bf16* p) {
  v8bf lo = *(const v8bf*)(p);
  v8bf hi = *(const v8bf*)(p + 8);
  return __builtin_shufflevector(lo, hi, 0,1,2,3,4,5,6,7,8,9,10,11,12,13,14,15);
}

// ---- GEMM: out = X @ W^T + bias, all-bf16 operands, f32 accumulate ---------
// One wave computes a 32x64 output tile (2 M-subtiles x 4 N-subtiles).
// Grid: (Mn/32, En/64). B fragments are reused across both M-subtiles.
template <int MODE>
__global__ __launch_bounds__(32) __attribute__((amdgpu_waves_per_eu(1)))
void gemm_bias_kernel(const __bf16* __restrict__ X,
                      const __bf16* __restrict__ Wb,
                      const float* __restrict__ bias,
                      void* __restrict__ outv) {
  const int lane = threadIdx.x & 31;
  const int lh   = lane >> 4;   // lane half (0/1)
  const int l16  = lane & 15;
  const int m0 = blockIdx.x * 32;
  const int n0 = blockIdx.y * 64;

  v8f acc[2][4] = {};

  for (int k0 = 0; k0 < En; k0 += 32) {
    v16bf afrag[2];
#pragma unroll
    for (int mt = 0; mt < 2; ++mt)
      afrag[mt] = load_afrag(X + (size_t)(m0 + mt * 16 + l16) * En + k0 + lh * 16);
    if (k0 + 32 < En) {  // prefetch next k-slice of A rows
      __builtin_prefetch(X + (size_t)(m0 + l16) * En + k0 + 32, 0, 1);
      __builtin_prefetch(X + (size_t)(m0 + 16 + l16) * En + k0 + 32, 0, 1);
    }
#pragma unroll
    for (int nt = 0; nt < 4; ++nt) {
      // B fragment (32x16): lane = column, K = lh*16 + 0..15 contiguous bf16
      const v16bf bfrag = *(const v16bf*)(
          Wb + (size_t)(n0 + nt * 16 + l16) * En + k0 + lh * 16);
      acc[0][nt] = __builtin_amdgcn_wmma_f32_16x16x32_bf16(
          false, afrag[0], false, bfrag, (short)0, acc[0][nt], false, false);
      acc[1][nt] = __builtin_amdgcn_wmma_f32_16x16x32_bf16(
          false, afrag[1], false, bfrag, (short)0, acc[1][nt], false, false);
    }
  }

#pragma unroll
  for (int nt = 0; nt < 4; ++nt) {
    const int n = n0 + nt * 16 + l16;
    const float bb = bias[n];
#pragma unroll
    for (int mt = 0; mt < 2; ++mt) {
#pragma unroll
      for (int i = 0; i < 8; ++i) {
        const int m = m0 + mt * 16 + i + 8 * lh;  // C layout: lane=col, VGPR=row
        const float val = acc[mt][nt][i] + bb;
        if (MODE == MODE_F32) {
          ((float*)outv)[(size_t)m * En + n] = val;
        } else {
          const int hh = n >> 6, dd = n & 63;
          const int bi = m >> 11, ss = m & (Sn - 1);
          __bf16* ob = (__bf16*)outv;
          if (MODE == MODE_HEAD)
            ob[(((size_t)bi * Hn + hh) * Sn + ss) * DKn + dd] = (__bf16)val;
          else  // MODE_VT: V stored transposed [b,h,d,s]
            ob[(((size_t)bi * Hn + hh) * DKn + dd) * Sn + ss] = (__bf16)val;
        }
      }
    }
  }
}

// ---- flash attention: one wave per (b, h, 32-row q tile) -------------------
// 32 q-rows per wave: every K/V fragment and mask row pair feeds two WMMAs.
__global__ __launch_bounds__(32) __attribute__((amdgpu_waves_per_eu(1)))
void attention_kernel(const __bf16* __restrict__ q_ws,
                      const __bf16* __restrict__ k_ws,
                      const __bf16* __restrict__ v_ws,
                      const int* __restrict__ mask,
                      __bf16* __restrict__ x_ws) {
  __shared__ __align__(32) __bf16 psh[32 * 32];  // P tiles staging (C->A transpose)

  const int lane = threadIdx.x & 31;
  const int lh   = lane >> 4;
  const int l16  = lane & 15;
  const int s0 = blockIdx.x * 32;
  const int h  = blockIdx.y;
  const int b  = blockIdx.z;
  const size_t bh = (size_t)b * Hn + h;

  const __bf16* qp = q_ws + bh * Sn * DKn;     // [s][d]
  const __bf16* kp = k_ws + bh * Sn * DKn;     // [s][d]
  const __bf16* vp = v_ws + bh * DKn * Sn;     // [d][s] (transposed)

  // Q tiles as A fragments for QK^T (K-dim = d = 64 -> two 16x32 frags per tile)
  v16bf aQ[2][2];
#pragma unroll
  for (int mt = 0; mt < 2; ++mt)
#pragma unroll
    for (int c = 0; c < 2; ++c)
      aQ[mt][c] =
          load_afrag(qp + (size_t)(s0 + mt * 16 + l16) * DKn + c * 32 + lh * 16);

  float rm[2][8], rl[2][8];
#pragma unroll
  for (int mt = 0; mt < 2; ++mt)
#pragma unroll
    for (int i = 0; i < 8; ++i) { rm[mt][i] = -3.0e38f; rl[mt][i] = 0.0f; }
  v8f o[2][4] = {};

  for (int j = 0; j < Sn; j += 32) {
    if (j + 32 < Sn) {  // prefetch next key/value block (global_prefetch_b8)
      __builtin_prefetch(kp + (size_t)(j + 32 + l16) * DKn, 0, 1);
      __builtin_prefetch(vp + (size_t)l16 * Sn + j + 32, 0, 1);
    }

    // K fragments for 32 keys (shared by both q tiles)
    v16bf bK[2][2];
#pragma unroll
    for (int t = 0; t < 2; ++t)
#pragma unroll
      for (int cc = 0; cc < 2; ++cc)
        bK[t][cc] = *(const v16bf*)(
            kp + (size_t)(j + t * 16 + l16) * DKn + cc * 32 + lh * 16);

    // S = Q K^T -> four 16x16 f32 tiles (2 q-tiles x 2 key-subblocks)
    v8f st[2][2];
#pragma unroll
    for (int mt = 0; mt < 2; ++mt)
#pragma unroll
      for (int t = 0; t < 2; ++t) {
        v8f c = {};
#pragma unroll
        for (int cc = 0; cc < 2; ++cc)
          c = __builtin_amdgcn_wmma_f32_16x16x32_bf16(
              false, aQ[mt][cc], false, bK[t][cc], (short)0, c, false, false);
        st[mt][t] = c;
      }

    // scale + mask + online softmax (row reductions over 16-lane halves)
#pragma unroll
    for (int mt = 0; mt < 2; ++mt) {
#pragma unroll
      for (int i = 0; i < 8; ++i) {
        const int q = s0 + mt * 16 + i + 8 * lh;
        const int* mrow = mask + ((size_t)b * Sn + q) * Sn + j;
        float s0v = st[mt][0][i] * 0.125f;     // 1/sqrt(64)
        float s1v = st[mt][1][i] * 0.125f;
        if (mrow[l16] == 0)      s0v = -1.0e9f;
        if (mrow[16 + l16] == 0) s1v = -1.0e9f;

        float mx = fmaxf(s0v, s1v);
#pragma unroll
        for (int off = 8; off >= 1; off >>= 1)
          mx = fmaxf(mx, __shfl_xor(mx, off, 16));
        const float nm = fmaxf(rm[mt][i], mx);

        const float e0 = __expf(s0v - nm);
        const float e1 = __expf(s1v - nm);
        float ps = e0 + e1;
#pragma unroll
        for (int off = 8; off >= 1; off >>= 1)
          ps += __shfl_xor(ps, off, 16);

        const float fac = __expf(rm[mt][i] - nm);
        rl[mt][i] = rl[mt][i] * fac + ps;
        rm[mt][i] = nm;
#pragma unroll
        for (int dt = 0; dt < 4; ++dt) o[mt][dt][i] *= fac;

        // stage P (C layout: lane=col) into LDS row-major for A-frag reload
        psh[(mt * 16 + i + 8 * lh) * 32 + l16]      = (__bf16)e0;
        psh[(mt * 16 + i + 8 * lh) * 32 + 16 + l16] = (__bf16)e1;
      }
    }
    __syncthreads();

    v16bf aP[2];
#pragma unroll
    for (int mt = 0; mt < 2; ++mt)
      aP[mt] = load_afrag(psh + (size_t)(mt * 16 + l16) * 32 + lh * 16);

#pragma unroll
    for (int dt = 0; dt < 4; ++dt) {
      const v16bf bV = *(const v16bf*)(
          vp + (size_t)(dt * 16 + l16) * Sn + j + lh * 16);
      o[0][dt] = __builtin_amdgcn_wmma_f32_16x16x32_bf16(
          false, aP[0], false, bV, (short)0, o[0][dt], false, false);
      o[1][dt] = __builtin_amdgcn_wmma_f32_16x16x32_bf16(
          false, aP[1], false, bV, (short)0, o[1][dt], false, false);
    }
    __syncthreads();
  }

  // normalize and store merged-head output [b,s,E] as bf16
#pragma unroll
  for (int mt = 0; mt < 2; ++mt) {
#pragma unroll
    for (int i = 0; i < 8; ++i) {
      const float inv = 1.0f / rl[mt][i];
      const int srow = s0 + mt * 16 + i + 8 * lh;
#pragma unroll
      for (int dt = 0; dt < 4; ++dt) {
        x_ws[((size_t)b * Sn + srow) * En + h * DKn + dt * 16 + l16] =
            (__bf16)(o[mt][dt][i] * inv);
      }
    }
  }
}

}  // namespace

extern "C" void kernel_launch(void* const* d_in, const int* in_sizes, int n_in,
                              void* d_out, int out_size, void* d_ws, size_t ws_size,
                              hipStream_t stream) {
  const float* query = (const float*)d_in[0];
  const float* key   = (const float*)d_in[1];
  const float* value = (const float*)d_in[2];
  const int*   mask  = (const int*)d_in[3];
  const float* Wq = (const float*)d_in[4];
  const float* bq = (const float*)d_in[5];
  const float* Wk = (const float*)d_in[6];
  const float* bk = (const float*)d_in[7];
  const float* Wv = (const float*)d_in[8];
  const float* bv = (const float*)d_in[9];
  const float* Wo = (const float*)d_in[10];
  const float* bo = (const float*)d_in[11];

  const size_t NTOK = (size_t)Bn * Hn * Sn * DKn;  // 4,194,304 (8 MB bf16)
  const size_t NW   = (size_t)En * En;             // 1,048,576 (2 MB bf16)
  const size_t NACT = (size_t)Mn * En;             // 4,194,304 (8 MB bf16)

  __bf16* p = (__bf16*)d_ws;
  __bf16* q_ws = p;            p += NTOK;
  __bf16* k_ws = p;            p += NTOK;
  __bf16* v_ws = p;            p += NTOK;
  __bf16* x_ws = p;            p += NTOK;
  __bf16* wq_b = p;            p += NW;
  __bf16* wk_b = p;            p += NW;
  __bf16* wv_b = p;            p += NW;
  __bf16* wo_b = p;            p += NW;
  __bf16* qi_b = p;            p += NACT;
  __bf16* ki_b = p;            p += NACT;
  __bf16* vi_b = p;            p += NACT;

  // 1) one-shot f32 -> bf16 conversion of weights and activations
  {
    const int nw8 = (int)(NW / 8), na8 = (int)(NACT / 8);
    const int tb = 256;
    cvt_bf16_kernel<<<(nw8 + tb - 1) / tb, tb, 0, stream>>>(Wq, wq_b, nw8);
    cvt_bf16_kernel<<<(nw8 + tb - 1) / tb, tb, 0, stream>>>(Wk, wk_b, nw8);
    cvt_bf16_kernel<<<(nw8 + tb - 1) / tb, tb, 0, stream>>>(Wv, wv_b, nw8);
    cvt_bf16_kernel<<<(nw8 + tb - 1) / tb, tb, 0, stream>>>(Wo, wo_b, nw8);
    cvt_bf16_kernel<<<(na8 + tb - 1) / tb, tb, 0, stream>>>(query, qi_b, na8);
    cvt_bf16_kernel<<<(na8 + tb - 1) / tb, tb, 0, stream>>>(key,   ki_b, na8);
    cvt_bf16_kernel<<<(na8 + tb - 1) / tb, tb, 0, stream>>>(value, vi_b, na8);
  }

  // 2) projections (32x64 output tile per wave)
  dim3 gproj(Mn / 32, En / 64, 1);
  gemm_bias_kernel<MODE_HEAD><<<gproj, 32, 0, stream>>>(qi_b, wq_b, bq, q_ws);
  gemm_bias_kernel<MODE_HEAD><<<gproj, 32, 0, stream>>>(ki_b, wk_b, bk, k_ws);
  gemm_bias_kernel<MODE_VT  ><<<gproj, 32, 0, stream>>>(vi_b, wv_b, bv, v_ws);

  // 3) flash attention (32 q-rows per wave)
  attention_kernel<<<dim3(Sn / 32, Hn, Bn), 32, 0, stream>>>(q_ws, k_ws, v_ws, mask, x_ws);

  // 4) output projection, f32 result to d_out
  gemm_bias_kernel<MODE_F32><<<gproj, 32, 0, stream>>>(x_ws, wo_b, bo, d_out);
}
